// GANetwork_59193239273551
// MI455X (gfx1250) — matrix operand order repaired and checked by reading, not compiled, not run on hardware
//
#include <hip/hip_runtime.h>
#include <hip/hip_bf16.h>

#define MDIM   4096
#define IN_F   512
#define OUT_F  64
#define HEADS  4
#define HO     (OUT_F * HEADS)   // 256
#define NCLS   40
#define NCLS_P 48                // padded to 3 N-tiles
#define NEG_SLOPE 0.01f
#define SPLITS 4                 // flash split-K over the j (neighbor) dimension
#define JSEG   (MDIM / SPLITS)   // 1024

typedef _Float16 v16h __attribute__((ext_vector_type(16)));
typedef _Float16 v8h  __attribute__((ext_vector_type(8)));
typedef float    v8f  __attribute__((ext_vector_type(8)));

union AV16 { v16h v; v8h h[2]; _Float16 e[16]; };

// A-fragment (16x32 f16): lane (m = lane&15, grp = lane>>4) reads two contiguous
// 8-half runs at K offsets 8*grp and 16+8*grp.
__device__ __forceinline__ void load_fragA(const _Float16* row, int k, int grp, AV16& f) {
    const _Float16* p = row + k + 8 * grp;
    f.h[0] = *(const v8h*)(p);
    f.h[1] = *(const v8h*)(p + 16);
}
// B-fragment (32x16 f16) from [N][K] storage: lane n=lane&15 reads two contiguous
// 8-half runs at K offsets 8*grp and 16+8*grp.
__device__ __forceinline__ void load_fragB(const _Float16* row, int k, int grp, AV16& f) {
    const _Float16* p = row + k + 8 * grp;
    f.h[0] = *(const v8h*)(p);
    f.h[1] = *(const v8h*)(p + 16);
}

// ---------------------------------------------------------------------------
// f32 -> f16 conversion with zero padding
// ---------------------------------------------------------------------------
__global__ void cvt_f16_kernel(const float* __restrict__ src, _Float16* __restrict__ dst,
                               int n, int ntot) {
    int i = blockIdx.x * blockDim.x + threadIdx.x;
    if (i < ntot) dst[i] = (i < n) ? (_Float16)src[i] : (_Float16)0.0f;
}

// ---------------------------------------------------------------------------
// WMMA f16 GEMM:  C[M][N] = A[M][K](f16 row-major) @ B[N][K](f16 weight layout)^T + bias
// One wave computes a 16 x (16*NT) tile; k-loop double-buffered.
// ---------------------------------------------------------------------------
template <int NT>
__global__ __launch_bounds__(32) void gemm_wmma_kernel(
    const _Float16* __restrict__ A, int lda,
    const _Float16* __restrict__ B, int ldb,
    const float* __restrict__ bias,
    int K, int Nvalid,
    float* __restrict__ out32, int ldo,
    _Float16* __restrict__ out16, int ldo16,
    _Float16* __restrict__ outT, int ldT)
{
    const int lane = threadIdx.x & 31;
    const int m    = lane & 15;
    const int grp  = lane >> 4;
    const int mbase = blockIdx.x * 16;
    const int nbase = blockIdx.y * 64;

    v8f acc[NT];
    #pragma unroll
    for (int t = 0; t < NT; ++t)
        #pragma unroll
        for (int i = 0; i < 8; ++i) acc[t][i] = 0.0f;

    const _Float16* arow = A + (long)(mbase + m) * lda;
    const _Float16* brow[NT];
    #pragma unroll
    for (int t = 0; t < NT; ++t)
        brow[t] = B + (long)(nbase + t * 16 + m) * ldb;

    AV16 af, bf[NT];
    load_fragA(arow, 0, grp, af);
    #pragma unroll
    for (int t = 0; t < NT; ++t) load_fragB(brow[t], 0, grp, bf[t]);

    for (int k = 0; k < K; k += 32) {
        AV16 afn, bfn[NT];
        const int kn = k + 32;
        if (kn < K) {                       // prefetch next chunk before WMMAs
            load_fragA(arow, kn, grp, afn);
            #pragma unroll
            for (int t = 0; t < NT; ++t) load_fragB(brow[t], kn, grp, bfn[t]);
        }
        #pragma unroll
        for (int t = 0; t < NT; ++t)
            acc[t] = __builtin_amdgcn_wmma_f32_16x16x32_f16(
                false, af.v, false, bf[t].v, (short)0, acc[t], false, false);
        if (kn < K) {
            af = afn;
            #pragma unroll
            for (int t = 0; t < NT; ++t) bf[t] = bfn[t];
        }
    }

    #pragma unroll
    for (int t = 0; t < NT; ++t) {
        const int col = nbase + t * 16 + m;
        const float bv = (bias && col < Nvalid) ? bias[col] : 0.0f;
        #pragma unroll
        for (int r = 0; r < 8; ++r) {
            const int row = mbase + r + 8 * grp;
            const float v = acc[t][r] + bv;
            if (out32 && col < Nvalid) out32[(long)row * ldo + col] = v;
            if (out16) out16[(long)row * ldo16 + col] = (_Float16)v;
            if (outT)  outT [(long)col * ldT   + row] = (_Float16)v;
        }
    }
}

// ---------------------------------------------------------------------------
// Per-node attention scores: src[h][j] = sum_c WhT[h*64+c][j]*a[h][c]
//                            dst[h][j] = sum_c WhT[h*64+c][j]*a[h][64+c]
// ---------------------------------------------------------------------------
__global__ void scores_kernel(const _Float16* __restrict__ WhT,
                              const float* __restrict__ a,
                              float* __restrict__ srcO, float* __restrict__ dstO)
{
    const int j = blockIdx.x * blockDim.x + threadIdx.x;
    const int head = blockIdx.y;
    if (j >= MDIM) return;
    const _Float16* base = WhT + (long)(head * OUT_F) * MDIM + j;
    const float* ah = a + head * (2 * OUT_F);
    float s = 0.0f, d = 0.0f;
    #pragma unroll 4
    for (int c = 0; c < OUT_F; ++c) {
        const float w = (float)base[(long)c * MDIM];
        s += w * ah[c];
        d += w * ah[OUT_F + c];
    }
    srcO[head * MDIM + j] = s;
    dstO[head * MDIM + j] = d;
}

// ---------------------------------------------------------------------------
// Flash attention PARTIAL: one wave per (16-row block, head, j-split of 1024).
// Online softmax over the segment; unnormalized accumulator + per-row (m,s)
// stats stored to workspace. 4x parallelism vs monolithic version.
// ---------------------------------------------------------------------------
__global__ __launch_bounds__(32) void attn_partial_kernel(
    const int* __restrict__ Amask,          // [MDIM][MDIM] 0/1
    const float* __restrict__ src,          // [HEADS][MDIM]
    const float* __restrict__ dst,          // [HEADS][MDIM]
    const _Float16* __restrict__ WhT,       // [HO][MDIM]
    float* __restrict__ pacc,               // [ib][head][split][16][64]
    float* __restrict__ pms)                // [ib][head][split][16][2]
{
    const int lane  = threadIdx.x & 31;
    const int m     = lane & 15;
    const int grp   = lane >> 4;
    const int ib    = blockIdx.x * 16;
    const int head  = blockIdx.y;
    const int split = blockIdx.z;
    const int jstart = split * JSEG;
    const float NEG_INF = -__builtin_huge_valf();

    const float srcv = src[head * MDIM + ib + m];
    const int*   arow = Amask + (long)(ib + m) * MDIM;
    const float* drow = dst + head * MDIM;
    const _Float16* Wb[4];
    #pragma unroll
    for (int t = 0; t < 4; ++t)
        Wb[t] = WhT + (long)(head * OUT_F + t * 16 + m) * MDIM;

    float mrun = NEG_INF, srun = 0.0f;
    v8f acc[4];
    #pragma unroll
    for (int t = 0; t < 4; ++t)
        #pragma unroll
        for (int i = 0; i < 8; ++i) acc[t][i] = 0.0f;

    for (int jb = jstart; jb < jstart + JSEG; jb += 32) {
        const int j0 = jb + 8 * grp;   // halves 0..7 -> j0.., halves 8..15 -> j0+16..

        // ---- issue ALL loads for this chunk up front ----
        int mk[16]; float dv[16];
        *(int4*)&mk[0]    = *(const int4*)(arow + j0);
        *(int4*)&mk[4]    = *(const int4*)(arow + j0 + 4);
        *(int4*)&mk[8]    = *(const int4*)(arow + j0 + 16);
        *(int4*)&mk[12]   = *(const int4*)(arow + j0 + 20);
        *(float4*)&dv[0]  = *(const float4*)(drow + j0);
        *(float4*)&dv[4]  = *(const float4*)(drow + j0 + 4);
        *(float4*)&dv[8]  = *(const float4*)(drow + j0 + 16);
        *(float4*)&dv[12] = *(const float4*)(drow + j0 + 20);
        AV16 bf[4];
        #pragma unroll
        for (int t = 0; t < 4; ++t) load_fragB(Wb[t], jb, grp, bf[t]);

        // ---- softmax VALU work (overlaps the Wh fragment loads) ----
        float l[16];
        float lmax = NEG_INF;
        #pragma unroll
        for (int h = 0; h < 16; ++h) {
            float s = srcv + dv[h];
            s = (s > 0.0f) ? s : NEG_SLOPE * s;       // leaky_relu BEFORE mask (matches ref)
            l[h] = (mk[h] != 0) ? s : NEG_INF;
            lmax = fmaxf(lmax, l[h]);
        }
        lmax = fmaxf(lmax, __shfl_xor(lmax, 16));     // combine the 2 lanes of each row
        const float mnew  = fmaxf(mrun, lmax);
        const float scale = (mrun == mnew) ? 1.0f : __expf(mrun - mnew);

        AV16 pf;
        float ls = 0.0f;
        #pragma unroll
        for (int h = 0; h < 16; ++h) {
            const float p = (l[h] == NEG_INF) ? 0.0f : __expf(l[h] - mnew);
            pf.e[h] = (_Float16)p;
            ls += p;
        }
        ls += __shfl_xor(ls, 16);
        srun = srun * scale + ls;
        mrun = mnew;

        // broadcast per-row rescale into C-fragment layout (VGPR r <-> row r+8*grp)
        float sc[8];
        #pragma unroll
        for (int r = 0; r < 8; ++r) sc[r] = __shfl(scale, r + 8 * grp);
        #pragma unroll
        for (int t = 0; t < 4; ++t)
            #pragma unroll
            for (int r = 0; r < 8; ++r) acc[t][r] *= sc[r];

        #pragma unroll
        for (int t = 0; t < 4; ++t)
            acc[t] = __builtin_amdgcn_wmma_f32_16x16x32_f16(
                false, pf.v, false, bf[t].v, (short)0, acc[t], false, false);
    }

    // ---- store unnormalized partial + per-row stats ----
    const long base = ((long)blockIdx.x * HEADS + head) * SPLITS + split;
    float* pa = pacc + base * (16 * 64);
    #pragma unroll
    for (int t = 0; t < 4; ++t)
        #pragma unroll
        for (int r = 0; r < 8; ++r)
            pa[(r + 8 * grp) * 64 + t * 16 + m] = acc[t][r];
    if (lane < 16) {                 // lanes 0..15 hold full stats for row = lane
        pms[base * 32 + lane * 2 + 0] = mrun;
        pms[base * 32 + lane * 2 + 1] = srun;
    }
}

// ---------------------------------------------------------------------------
// Combine SPLITS partials: out[row][col] = sum_p acc_p*exp(m_p-mg) / sum_p s_p*exp(m_p-mg)
// ---------------------------------------------------------------------------
__global__ void attn_combine_kernel(
    const float* __restrict__ pacc, const float* __restrict__ pms,
    float* __restrict__ out32, int ldo,
    _Float16* __restrict__ out16, int ldo16)
{
    const int ib   = blockIdx.x;
    const int head = blockIdx.y;
    const float NEG_INF = -__builtin_huge_valf();
    const long base = ((long)ib * HEADS + head) * SPLITS;

    for (int e = threadIdx.x; e < 16 * 64; e += blockDim.x) {
        const int row = e >> 6;
        const int col = e & 63;
        float mg = NEG_INF;
        #pragma unroll
        for (int p = 0; p < SPLITS; ++p)
            mg = fmaxf(mg, pms[(base + p) * 32 + row * 2]);
        float stot = 0.0f, val = 0.0f;
        #pragma unroll
        for (int p = 0; p < SPLITS; ++p) {
            const float mp = pms[(base + p) * 32 + row * 2 + 0];
            const float sp = pms[(base + p) * 32 + row * 2 + 1];
            const float w  = (mp == NEG_INF) ? 0.0f : __expf(mp - mg);
            stot += sp * w;
            val  += pacc[(base + p) * (16 * 64) + e] * w;
        }
        const float o = (stot > 0.0f) ? val / stot : 0.0f;
        const int orow = ib * 16 + row;
        const int ocol = head * OUT_F + col;
        if (out32) out32[(long)orow * ldo + ocol] = o;
        out16[(long)orow * ldo16 + ocol] = (_Float16)o;
    }
}

// ---------------------------------------------------------------------------
extern "C" void kernel_launch(void* const* d_in, const int* in_sizes, int n_in,
                              void* d_out, int out_size, void* d_ws, size_t ws_size,
                              hipStream_t stream) {
    const float* h    = (const float*)d_in[0];
    const int*   A    = (const int*)  d_in[1];
    const float* W1_w = (const float*)d_in[2];
    const float* W1_b = (const float*)d_in[3];
    const float* a1   = (const float*)d_in[4];
    const float* W3_w = (const float*)d_in[5];
    const float* W3_b = (const float*)d_in[6];
    const float* W2_w = (const float*)d_in[7];
    const float* W2_b = (const float*)d_in[8];
    const float* a2   = (const float*)d_in[9];
    const float* FL_w = (const float*)d_in[10];
    const float* FL_b = (const float*)d_in[11];

    float* x2_out  = (float*)d_out;                          // (4096, 256)
    float* cls_out = (float*)d_out + (size_t)MDIM * HO;      // (4096, 40)

    // bump allocator over workspace
    char* wp = (char*)d_ws;
    auto alloc = [&](size_t bytes) -> void* {
        void* r = (void*)wp;
        wp += (bytes + 255) & ~(size_t)255;
        return r;
    };
    _Float16* h16   = (_Float16*)alloc(sizeof(_Float16) * (size_t)MDIM * IN_F);
    _Float16* W1f   = (_Float16*)alloc(sizeof(_Float16) * (size_t)HO * IN_F);
    _Float16* W3f   = (_Float16*)alloc(sizeof(_Float16) * (size_t)OUT_F * HO);
    _Float16* W2f   = (_Float16*)alloc(sizeof(_Float16) * (size_t)HO * OUT_F);
    _Float16* FLf   = (_Float16*)alloc(sizeof(_Float16) * (size_t)NCLS_P * HO);
    _Float16* WhT1  = (_Float16*)alloc(sizeof(_Float16) * (size_t)HO * MDIM);
    _Float16* WhT2  = (_Float16*)alloc(sizeof(_Float16) * (size_t)HO * MDIM);
    _Float16* x1f   = (_Float16*)alloc(sizeof(_Float16) * (size_t)MDIM * HO);
    _Float16* xf    = (_Float16*)alloc(sizeof(_Float16) * (size_t)MDIM * OUT_F);
    _Float16* x2f   = (_Float16*)alloc(sizeof(_Float16) * (size_t)MDIM * HO);
    float*    src1  = (float*)alloc(sizeof(float) * HEADS * MDIM);
    float*    dst1  = (float*)alloc(sizeof(float) * HEADS * MDIM);
    float*    src2  = (float*)alloc(sizeof(float) * HEADS * MDIM);
    float*    dst2  = (float*)alloc(sizeof(float) * HEADS * MDIM);
    float*    pacc  = (float*)alloc(sizeof(float) * (size_t)(MDIM / 16) * HEADS * SPLITS * 16 * 64);
    float*    pms   = (float*)alloc(sizeof(float) * (size_t)(MDIM / 16) * HEADS * SPLITS * 32);

    auto cvt = [&](const float* s, _Float16* d, int n, int ntot) {
        cvt_f16_kernel<<<dim3((ntot + 255) / 256), dim3(256), 0, stream>>>(s, d, n, ntot);
    };
    cvt(h,    h16, MDIM * IN_F,  MDIM * IN_F);
    cvt(W1_w, W1f, HO * IN_F,    HO * IN_F);
    cvt(W3_w, W3f, OUT_F * HO,   OUT_F * HO);
    cvt(W2_w, W2f, HO * OUT_F,   HO * OUT_F);
    cvt(FL_w, FLf, NCLS * HO,    NCLS_P * HO);   // zero-padded rows 40..47

    // GEMM1: Wh1 = h @ W1^T + b1  -> Wh1^T f16 [256][4096]
    gemm_wmma_kernel<4><<<dim3(MDIM / 16, HO / 64), dim3(32), 0, stream>>>(
        h16, IN_F, W1f, IN_F, W1_b, IN_F, HO,
        nullptr, 0, nullptr, 0, WhT1, MDIM);

    scores_kernel<<<dim3(MDIM / 256, HEADS), dim3(256), 0, stream>>>(WhT1, a1, src1, dst1);

    // attention 1 (split-j) -> x1 f16 row-major
    attn_partial_kernel<<<dim3(MDIM / 16, HEADS, SPLITS), dim3(32), 0, stream>>>(
        A, src1, dst1, WhT1, pacc, pms);
    attn_combine_kernel<<<dim3(MDIM / 16, HEADS), dim3(256), 0, stream>>>(
        pacc, pms, nullptr, 0, x1f, HO);

    // GEMM3: x = x1 @ W3^T + b3 -> f16 row-major (4096, 64)
    gemm_wmma_kernel<4><<<dim3(MDIM / 16, 1), dim3(32), 0, stream>>>(
        x1f, HO, W3f, HO, W3_b, HO, OUT_F,
        nullptr, 0, xf, OUT_F, nullptr, 0);

    // GEMM2: Wh2 = x @ W2^T + b2 -> Wh2^T f16 [256][4096]
    gemm_wmma_kernel<4><<<dim3(MDIM / 16, HO / 64), dim3(32), 0, stream>>>(
        xf, OUT_F, W2f, OUT_F, W2_b, OUT_F, HO,
        nullptr, 0, nullptr, 0, WhT2, MDIM);

    scores_kernel<<<dim3(MDIM / 256, HEADS), dim3(256), 0, stream>>>(WhT2, a2, src2, dst2);

    // attention 2 (split-j) -> x2: f32 into d_out + f16 copy for classifier
    attn_partial_kernel<<<dim3(MDIM / 16, HEADS, SPLITS), dim3(32), 0, stream>>>(
        A, src2, dst2, WhT2, pacc, pms);
    attn_combine_kernel<<<dim3(MDIM / 16, HEADS), dim3(256), 0, stream>>>(
        pacc, pms, x2_out, HO, x2f, HO);

    // Final: out = x2 @ FL^T + b  -> f32 (4096, 40), padded N=48, stores guarded
    gemm_wmma_kernel<3><<<dim3(MDIM / 16, 1), dim3(32), 0, stream>>>(
        x2f, HO, FLf, HO, FL_b, HO, NCLS,
        cls_out, NCLS, nullptr, 0, nullptr, 0);
}
